// SCPMambaLayer_60232621359316
// MI455X (gfx1250) — matrix operand order repaired and checked by reading, not compiled, not run on hardware
//
#include <hip/hip_runtime.h>

#define H96   96
#define W96   96
#define HWSZ  9216
#define BATCH 4
#define MTOT  36864   // BATCH * HWSZ
#define LSEQ  9216
#define CCAT  160

typedef __attribute__((ext_vector_type(16))) _Float16 v16h;
typedef __attribute__((ext_vector_type(8)))  float    v8f;

enum { MODE_PLAIN = 0, MODE_NCHW = 1, MODE_MC = 2 };

#define CVT4(dst, base, p)                      \
    dst[(base) + 0] = (_Float16)(p).x;          \
    dst[(base) + 1] = (_Float16)(p).y;          \
    dst[(base) + 2] = (_Float16)(p).z;          \
    dst[(base) + 3] = (_Float16)(p).w;

// ---------------------------------------------------------------------------
// LDS-free WMMA GEMM: C[m, cofs+n] = act( sum_k A_im2col[m,k] * Wt[n,k] + bias[n] )
//                     optionally  += ss[0] * resid[m, cofs+n]
// Fragments built directly from global memory in contiguous k order (same for A and B):
//   A 16x32 : lane(mr,hi) holds row m0+mr, K = k0+hi*16+j  (16 contiguous k)
//   B 32x16 : lane(mr,hi) holds col n0+mr, K = k0+hi*16+j  (16 contiguous k)
//   C 16x16 : lane(mr,hi) holds col n0+mr, rows m0 + r + 8*hi
// OOB k contributes 0 via the A fragment; OOB n columns are never stored, so
// B loads only need address clamping, no value masking.
// MT M-tiles per wave share each B fragment (MT=2 for PLAIN GEMMs).
// Requirements: K % 16 == 0 always; K % 32 == 0 for MODE_PLAIN (true here).
// ---------------------------------------------------------------------------
template<int MODE, int KH, int KW, int MT>
__global__ void wmma_gemm_kernel(const float* __restrict__ A, int lda, int aofs,
                                 int Cin, int pad,
                                 const float* __restrict__ Wt, int K,
                                 const float* __restrict__ bias,
                                 float* __restrict__ Cout, int ldc, int cofs, int N,
                                 int relu,
                                 const float* __restrict__ resid,
                                 const float* __restrict__ ss)
{
    const int lane = threadIdx.x & 31;
    const int wv   = threadIdx.x >> 5;
    const int m0   = blockIdx.x * (16 * MT);
    const int n0   = (blockIdx.y * 4 + wv) << 4;
    const int mr   = lane & 15;
    const int hi   = lane >> 4;

    // B: this lane always reads row nn of Wt (clamped so the load is safe)
    const int  nn     = n0 + mr;
    const bool nvalid = nn < N;
    const float* brow = Wt + (size_t)(nvalid ? nn : 0) * K;

    // A: per-M-tile fragment rows
    int b_[MT], h_[MT], w_[MT];
    const float* arow[MT];
    #pragma unroll
    for (int t = 0; t < MT; ++t) {
        const int am = m0 + t * 16 + mr;
        if (MODE != MODE_PLAIN) {
            b_[t] = am / HWSZ;
            int hw = am - b_[t] * HWSZ;
            h_[t] = hw / W96;
            w_[t] = hw - h_[t] * W96;
        }
        arow[t] = A + (size_t)am * lda + aofs;     // used only in MODE_PLAIN
    }

    v8f acc[MT] = {};
    for (int k0 = 0; k0 < K; k0 += 32) {
        // ---------------- B fragment (16 contiguous k of one weight row) ----
        v16h bf;
        {
            const bool kv = (k0 + hi * 16) < K;    // clamp only; no masking needed
            const int  kb = kv ? (k0 + hi * 16) : 0;
            const float4 p0 = *(const float4*)(brow + kb);
            const float4 p1 = *(const float4*)(brow + kb + 4);
            const float4 p2 = *(const float4*)(brow + kb + 8);
            const float4 p3 = *(const float4*)(brow + kb + 12);
            CVT4(bf, 0,  p0) CVT4(bf, 4,  p1) CVT4(bf, 8,  p2) CVT4(bf, 12, p3)
        }
        // ---------------- A fragments + WMMA per M-tile ----------------
        #pragma unroll
        for (int t = 0; t < MT; ++t) {
            v16h af;
            if (MODE == MODE_PLAIN) {
                const float* ar = arow[t];
                const float4 q0 = *(const float4*)(ar + k0 + hi * 16);
                const float4 q1 = *(const float4*)(ar + k0 + hi * 16 + 4);
                const float4 q2 = *(const float4*)(ar + k0 + hi * 16 + 8);
                const float4 q3 = *(const float4*)(ar + k0 + hi * 16 + 12);
                CVT4(af, 0,  q0) CVT4(af, 4,  q1) CVT4(af, 8,  q2) CVT4(af, 12, q3)
            } else {
                #pragma unroll
                for (int j = 0; j < 16; ++j) {
                    const int k = k0 + hi * 16 + j;
                    int c   = k / (KH * KW);
                    int rem = k - c * (KH * KW);
                    int dh  = rem / KW;
                    int dw  = rem - dh * KW;
                    int ih  = h_[t] + dh - pad;
                    int iw  = w_[t] + dw - pad;
                    bool ok = (k < K) & (ih >= 0) & (ih < H96) & (iw >= 0) & (iw < W96);
                    int cih = ok ? ih : 0;
                    int ciw = ok ? iw : 0;
                    int ck  = ok ? c  : 0;
                    float v;
                    if (MODE == MODE_NCHW)
                        v = A[(((size_t)b_[t] * Cin + ck) * H96 + cih) * W96 + ciw];
                    else
                        v = A[(size_t)(b_[t] * HWSZ + cih * W96 + ciw) * Cin + ck];
                    af[j] = (_Float16)(ok ? v : 0.f);
                }
            }
            acc[t] = __builtin_amdgcn_wmma_f32_16x16x32_f16(false, af, false, bf,
                                                            (short)0, acc[t], false, false);
        }
    }
    // ---------------- store C ----------------
    if (nvalid) {
        float bi  = bias  ? bias[nn] : 0.f;
        float ssv = resid ? ss[0]    : 0.f;
        #pragma unroll
        for (int t = 0; t < MT; ++t) {
            #pragma unroll
            for (int r = 0; r < 8; ++r) {
                int mm   = m0 + t * 16 + r + (hi << 3);
                size_t o = (size_t)mm * ldc + cofs + nn;
                float v  = acc[t][r] + bi;
                if (relu)  v = fmaxf(v, 0.f);
                if (resid) v += ssv * resid[o];
                Cout[o] = v;
            }
        }
    }
}

// ---------------------------------------------------------------------------
__global__ void maxpool3_kernel(const float* __restrict__ x, float* __restrict__ out)
{
    int idx = blockIdx.x * blockDim.x + threadIdx.x;
    const int total = BATCH * 64 * HWSZ;
    if (idx >= total) return;
    int hw = idx % HWSZ;
    int bc = idx / HWSZ;
    int h = hw / W96, w = hw % W96;
    float m = -3.402823466e38f;
    #pragma unroll
    for (int dh = -1; dh <= 1; ++dh)
        #pragma unroll
        for (int dw = -1; dw <= 1; ++dw) {
            int ih = h + dh, iw = w + dw;
            if (ih >= 0 && ih < H96 && iw >= 0 && iw < W96)
                m = fmaxf(m, x[(size_t)bc * HWSZ + ih * W96 + iw]);
        }
    out[idx] = m;
}

// per-branch LayerNorm over C channels, scatter into XLN[m,160] at cofs
__global__ void ln_branch_kernel(const float* __restrict__ X, int C,
                                 const float* __restrict__ g, const float* __restrict__ b,
                                 float* __restrict__ out, int cofs)
{
    int m = blockIdx.x * blockDim.x + threadIdx.x;
    if (m >= MTOT) return;
    const float* r = X + (size_t)m * C;
    float mean = 0.f;
    for (int c = 0; c < C; ++c) mean += r[c];
    mean /= (float)C;
    float var = 0.f;
    for (int c = 0; c < C; ++c) { float t = r[c] - mean; var += t * t; }
    var /= (float)C;
    float inv = rsqrtf(var + 1e-5f);
    for (int c = 0; c < C; ++c)
        out[(size_t)m * CCAT + cofs + c] = (r[c] - mean) * inv * g[c] + b[c];
}

// causal depthwise conv1d (k=4, left pad 3) + SiLU:  U[m,d]
__global__ void dwconv_silu_kernel(const float* __restrict__ xz,
                                   const float* __restrict__ cw,
                                   const float* __restrict__ cb,
                                   float* __restrict__ U, int di)
{
    int idx = blockIdx.x * blockDim.x + threadIdx.x;
    int total = MTOT * di;
    if (idx >= total) return;
    int d = idx % di;
    int m = idx / di;
    int b = m / LSEQ;
    int l = m - b * LSEQ;
    float s = cb[d];
    #pragma unroll
    for (int j = 0; j < 4; ++j) {
        int l2 = l - 3 + j;
        if (l2 >= 0)
            s += cw[d * 4 + j] * xz[(size_t)(b * LSEQ + l2) * (2 * di) + d];
    }
    U[idx] = s / (1.f + __expf(-s));   // silu
}

// dt = softplus(dt_proj_w @ x_dbl[:, :dtr] + dt_proj_b)
__global__ void dtproj_kernel(const float* __restrict__ xdbl,
                              const float* __restrict__ dtw,
                              const float* __restrict__ dtb,
                              float* __restrict__ DT, int di, int dtr, int nx)
{
    int idx = blockIdx.x * blockDim.x + threadIdx.x;
    int total = MTOT * di;
    if (idx >= total) return;
    int d = idx % di;
    int m = idx / di;
    float s = dtb[d];
    for (int r = 0; r < dtr; ++r)
        s += xdbl[(size_t)m * nx + r] * dtw[d * dtr + r];
    DT[idx] = (s > 20.f) ? s : log1pf(__expf(s));
}

// selective scan: one thread per (batch, channel), 16-wide state in registers
__global__ void mamba_scan_kernel(const float* __restrict__ xz,
                                  const float* __restrict__ u,
                                  const float* __restrict__ xdbl,
                                  const float* __restrict__ dtv,
                                  const float* __restrict__ A_log,
                                  const float* __restrict__ Dp,
                                  float* __restrict__ Y,
                                  int di, int dtr, int nx)
{
    int b = blockIdx.x;
    int d = threadIdx.x;
    if (d >= di) return;
    float a[16], h[16];
    #pragma unroll
    for (int n = 0; n < 16; ++n) { a[n] = -__expf(A_log[d * 16 + n]); h[n] = 0.f; }
    float Dd = Dp[d];
    for (int l = 0; l < LSEQ; ++l) {
        size_t m = (size_t)b * LSEQ + l;
        float dt = dtv[m * di + d];
        float uu = u[m * di + d];
        const float* xb = xdbl + m * nx + dtr;    // B at [0..15], C at [16..31]
        float du = dt * uu;
        float acc = 0.f;
        #pragma unroll
        for (int n = 0; n < 16; ++n) {
            float dA = __expf(dt * a[n]);
            h[n] = dA * h[n] + du * xb[n];
            acc += h[n] * xb[16 + n];
        }
        float z = xz[m * (2 * di) + di + d];
        float sil = z / (1.f + __expf(-z));
        Y[m * di + d] = (acc + uu * Dd) * sil;
    }
}

// final: LN over 160-ch concat rows, transpose to NCHW output
__global__ void final_ln_kernel(const float* __restrict__ MC,
                                const float* __restrict__ g,
                                const float* __restrict__ b,
                                float* __restrict__ out)
{
    int m = blockIdx.x * blockDim.x + threadIdx.x;
    if (m >= MTOT) return;
    const float* r = MC + (size_t)m * CCAT;
    float mean = 0.f;
    for (int c = 0; c < CCAT; ++c) mean += r[c];
    mean *= (1.f / CCAT);
    float var = 0.f;
    for (int c = 0; c < CCAT; ++c) { float t = r[c] - mean; var += t * t; }
    var *= (1.f / CCAT);
    float inv = rsqrtf(var + 1e-5f);
    int bb = m / HWSZ, hw = m - bb * HWSZ;
    for (int c = 0; c < CCAT; ++c)
        out[((size_t)bb * CCAT + c) * HWSZ + hw] = (r[c] - mean) * inv * g[c] + b[c];
}

// ---------------------------------------------------------------------------
static inline dim3 gemm_grid1(int N) { return dim3(MTOT / 16, (N + 63) / 64); }
static inline dim3 gemm_grid2(int N) { return dim3(MTOT / 32, (N + 63) / 64); }

static void run_mamba(const float* const* P, int d, int cofs,
                      float* XLN, float* MC,
                      float* XZ, float* U, float* XDBL, float* DT, float* Y,
                      const float* ss, hipStream_t stream)
{
    const int di  = 2 * d;
    const int dtr = (d + 15) / 16;
    const int nx  = dtr + 32;
    const float* inw = P[0]; const float* cw  = P[1]; const float* cb  = P[2];
    const float* xpw = P[3]; const float* dtw = P[4]; const float* dtb = P[5];
    const float* alg = P[6]; const float* Dp  = P[7]; const float* opw = P[8];
    const int tot = MTOT * di;

    // in_proj: [M,d] x [2di,d]^T -> XZ [M, 2di]
    wmma_gemm_kernel<MODE_PLAIN, 1, 1, 2><<<gemm_grid2(2 * di), 128, 0, stream>>>(
        XLN, CCAT, cofs, 0, 0, inw, d, nullptr, XZ, 2 * di, 0, 2 * di, 0, nullptr, nullptr);
    dwconv_silu_kernel<<<(tot + 255) / 256, 256, 0, stream>>>(XZ, cw, cb, U, di);
    // x_proj: [M,di] x [nx,di]^T -> XDBL [M, nx]
    wmma_gemm_kernel<MODE_PLAIN, 1, 1, 2><<<gemm_grid2(nx), 128, 0, stream>>>(
        U, di, 0, 0, 0, xpw, di, nullptr, XDBL, nx, 0, nx, 0, nullptr, nullptr);
    dtproj_kernel<<<(tot + 255) / 256, 256, 0, stream>>>(XDBL, dtw, dtb, DT, di, dtr, nx);
    mamba_scan_kernel<<<dim3(BATCH), dim3(di), 0, stream>>>(XZ, U, XDBL, DT, alg, Dp, Y, di, dtr, nx);
    // out_proj + skip: MC[:,cofs:cofs+d] = Y @ opw^T + ss*XLN[:,cofs:cofs+d]
    wmma_gemm_kernel<MODE_PLAIN, 1, 1, 2><<<gemm_grid2(d), 128, 0, stream>>>(
        Y, di, 0, 0, 0, opw, di, nullptr, MC, CCAT, cofs, d, 0, XLN, ss);
}

extern "C" void kernel_launch(void* const* d_in, const int* in_sizes, int n_in,
                              void* d_out, int out_size, void* d_ws, size_t ws_size,
                              hipStream_t stream)
{
    if (n_in < 60 || !d_ws) return;
    const float* x = (const float*)d_in[0];
    auto F = [&](int i) { return (const float*)d_in[i]; };

    // workspace layout (floats)
    float* W0 = (float*)d_ws;
    size_t o = 0;
    float* XLN  = W0 + o; o += (size_t)MTOT * CCAT;  // normalized branch concat
    float* MC   = W0 + o; o += (size_t)MTOT * CCAT;  // mamba outputs + skip
    float* POOL = W0 + o; o += (size_t)MTOT * 64;    // maxpool (NCHW)
    float* TMP  = W0 + o; o += (size_t)MTOT * 64;    // branch conv output
    float* T1   = W0 + o; o += (size_t)MTOT * 32;    // branch reduce conv output
    float* XZ   = W0 + o; o += (size_t)MTOT * 256;
    float* U    = W0 + o; o += (size_t)MTOT * 128;
    float* XDBL = W0 + o; o += (size_t)MTOT * 48;
    float* DT   = W0 + o; o += (size_t)MTOT * 128;
    float* Y    = W0 + o; o += (size_t)MTOT * 128;

    const float* ss = F(23);
    const int mthreads = (MTOT + 255) / 256;

    // ---- branch 1: 1x1 conv 64->32 + relu, LN -> XLN[:, 0:32]
    wmma_gemm_kernel<MODE_NCHW, 1, 1, 1><<<gemm_grid1(32), 128, 0, stream>>>(
        x, 0, 0, 64, 0, F(1), 64, F(2), TMP, 32, 0, 32, 1, nullptr, nullptr);
    ln_branch_kernel<<<mthreads, 256, 0, stream>>>(TMP, 32, F(13), F(14), XLN, 0);

    // ---- branch 2: 1x1 64->32, 3x3 32->64 pad1, LN -> XLN[:, 32:96]
    wmma_gemm_kernel<MODE_NCHW, 1, 1, 1><<<gemm_grid1(32), 128, 0, stream>>>(
        x, 0, 0, 64, 0, F(3), 64, F(4), T1, 32, 0, 32, 1, nullptr, nullptr);
    wmma_gemm_kernel<MODE_MC, 3, 3, 1><<<gemm_grid1(64), 128, 0, stream>>>(
        T1, 0, 0, 32, 1, F(5), 288, F(6), TMP, 64, 0, 64, 1, nullptr, nullptr);
    ln_branch_kernel<<<mthreads, 256, 0, stream>>>(TMP, 64, F(15), F(16), XLN, 32);

    // ---- branch 3: 1x1 64->16, 5x5 16->32 pad2, LN -> XLN[:, 96:128]
    wmma_gemm_kernel<MODE_NCHW, 1, 1, 1><<<gemm_grid1(16), 128, 0, stream>>>(
        x, 0, 0, 64, 0, F(7), 64, F(8), T1, 16, 0, 16, 1, nullptr, nullptr);
    wmma_gemm_kernel<MODE_MC, 5, 5, 1><<<gemm_grid1(32), 128, 0, stream>>>(
        T1, 0, 0, 16, 2, F(9), 400, F(10), TMP, 32, 0, 32, 1, nullptr, nullptr);
    ln_branch_kernel<<<mthreads, 256, 0, stream>>>(TMP, 32, F(17), F(18), XLN, 96);

    // ---- branch 4: maxpool3 pad1, 1x1 64->32, LN -> XLN[:, 128:160]
    {
        int tot = BATCH * 64 * HWSZ;
        maxpool3_kernel<<<(tot + 255) / 256, 256, 0, stream>>>(x, POOL);
    }
    wmma_gemm_kernel<MODE_NCHW, 1, 1, 1><<<gemm_grid1(32), 128, 0, stream>>>(
        POOL, 0, 0, 64, 0, F(11), 64, F(12), TMP, 32, 0, 32, 1, nullptr, nullptr);
    ln_branch_kernel<<<mthreads, 256, 0, stream>>>(TMP, 32, F(19), F(20), XLN, 128);

    // ---- mamba blocks (params in _mamba_params return order) ----
    const float* m1p[9]; const float* m2p[9]; const float* m3p[9]; const float* m4p[9];
    for (int j = 0; j < 9; ++j) {
        m1p[j] = F(24 + j); m2p[j] = F(33 + j); m3p[j] = F(42 + j); m4p[j] = F(51 + j);
    }
    run_mamba(m1p, 32, 0,   XLN, MC, XZ, U, XDBL, DT, Y, ss, stream);
    run_mamba(m2p, 64, 32,  XLN, MC, XZ, U, XDBL, DT, Y, ss, stream);
    run_mamba(m3p, 32, 96,  XLN, MC, XZ, U, XDBL, DT, Y, ss, stream);
    run_mamba(m4p, 32, 128, XLN, MC, XZ, U, XDBL, DT, Y, ss, stream);

    // ---- final LN over concat + NCHW transpose ----
    final_ln_kernel<<<mthreads, 256, 0, stream>>>(MC, F(21), F(22), (float*)d_out);
}
